// Net_23665269801495
// MI455X (gfx1250) — compile-verified, matrix operand order checked by reading, not run on hardware
//
#include <hip/hip_runtime.h>
#include <hip/hip_bf16.h>
#include <math.h>

typedef __bf16 bf16_t;
typedef bf16_t v16bf __attribute__((ext_vector_type(16)));
typedef float  v8f   __attribute__((ext_vector_type(8)));

#define N_NODES  1024
#define N_EDGES  2048
#define N_GRAPHS 64

// ---------------------------------------------------------------------------
// Pre-swizzle w2 (fp32, [KEDGE, M_IN_REAL*M_OUT]) into bf16 WMMA B-fragments.
// Fragment flat layout: ((chunk*NTILES + tile)*32 + lane)*16 + h
// lane<16 : n = lane,    h->kk = (h&7) + 16*(h>>3)
// lane>=16: n = lane-16, h->kk = (h&7) + 8 + 16*(h>>3)
// p = chunk*32 + kk ; i = p & (M_IN_PAD-1) ; k = p >> LG_MIN
// ---------------------------------------------------------------------------
template<int LG_MIN, int M_IN_PAD, int M_IN_REAL, int KEDGE, int M_OUT>
__global__ __launch_bounds__(256) void swizzle_w2_kernel(
    const float* __restrict__ w2, bf16_t* __restrict__ frag)
{
  constexpr int NTILES = M_OUT / 16;
  constexpr long NCHUNK = (long)(M_IN_PAD) * KEDGE / 32;
  const long total = NCHUNK * NTILES * 512;
  for (long tix = (long)blockIdx.x * blockDim.x + threadIdx.x; tix < total;
       tix += (long)gridDim.x * blockDim.x) {
    const int h    = (int)(tix & 15);
    const int lane = (int)((tix >> 4) & 31);
    const int tile = (int)((tix >> 9) % NTILES);
    const long c   = tix / (512L * NTILES);
    const int kk = (h & 7) + ((lane >> 4) << 3) + ((h >> 3) << 4);
    const long p = c * 32 + kk;
    const int i = (int)(p & (M_IN_PAD - 1));
    const int k = (int)(p >> LG_MIN);
    const int o = tile * 16 + (lane & 15);
    float v = 0.f;
    if (i < M_IN_REAL)
      v = w2[(size_t)k * ((size_t)M_IN_REAL * M_OUT) + (size_t)i * M_OUT + o];
    frag[tix] = (bf16_t)v;
  }
}

// b2 ([M_IN_REAL*M_OUT]) swizzled the same way; the "K" dim here is i itself.
template<int M_IN_PAD, int M_IN_REAL, int M_OUT>
__global__ __launch_bounds__(256) void swizzle_b2_kernel(
    const float* __restrict__ b2, bf16_t* __restrict__ frag)
{
  constexpr int NTILES = M_OUT / 16;
  constexpr int CB = M_IN_PAD / 32;
  const int total = CB * NTILES * 512;
  for (int tix = blockIdx.x * blockDim.x + threadIdx.x; tix < total;
       tix += gridDim.x * blockDim.x) {
    const int h    = tix & 15;
    const int lane = (tix >> 4) & 31;
    const int tile = (tix >> 9) % NTILES;
    const int c    = tix / (512 * NTILES);
    const int kk = (h & 7) + ((lane >> 4) << 3) + ((h >> 3) << 4);
    const int i  = c * 32 + kk;
    const int o  = tile * 16 + (lane & 15);
    frag[tix] = (bf16_t)((i < M_IN_REAL) ? b2[(size_t)i * M_OUT + o] : 0.f);
  }
}

// ---------------------------------------------------------------------------
// Edge MLP layer 1: r = relu(edge_attr @ w1 + b1), fp32 output [E, K]
// ---------------------------------------------------------------------------
__global__ __launch_bounds__(256) void edge_mlp1_kernel(
    const float* __restrict__ ea, const float* __restrict__ w1,
    const float* __restrict__ b1, float* __restrict__ rout, int E, int K)
{
  const int t = blockIdx.x * blockDim.x + threadIdx.x;
  if (t >= E * K) return;
  const int e = t / K, k = t % K;
  float a = b1[k];
#pragma unroll
  for (int j = 0; j < 4; ++j) a = fmaf(ea[e * 4 + j], w1[j * K + k], a);
  rout[t] = fmaxf(a, 0.f);
}

// Gather source-node features per edge, zero-padding feature dim to Dpad.
__global__ __launch_bounds__(256) void gather_pad_kernel(
    const float* __restrict__ h, const int* __restrict__ src,
    float* __restrict__ xs, int E, int Dreal, int Dpad)
{
  const int t = blockIdx.x * blockDim.x + threadIdx.x;
  if (t >= E * Dpad) return;
  const int e = t / Dpad, i = t % Dpad;
  xs[t] = (i < Dreal) ? h[(size_t)src[e] * Dreal + i] : 0.f;
}

__global__ __launch_bounds__(256) void zero_kernel(float* __restrict__ p, long n)
{
  const long t = (long)blockIdx.x * blockDim.x + threadIdx.x;
  if (t < n) p[t] = 0.f;
}

// ---------------------------------------------------------------------------
// Core: msg[e,:] = Z[e,:] @ B  built on the fly from rank-1 structure,
// scatter-added with fp32 global atomics into aggr[tgt[e],:].
// Block = 128 threads (4 waves), one 16-edge tile per block, K split by grid.y.
// ---------------------------------------------------------------------------
template<int LG_MIN, int M_IN_PAD, int KEDGE, int M_OUT, int KSPLIT>
__global__ __launch_bounds__(128) void zgemm_scatter_kernel(
    const bf16_t* __restrict__ Bfrag,   // swizzled w2
    const bf16_t* __restrict__ b2frag,  // swizzled b2
    const float*  __restrict__ xs,      // [E, M_IN_PAD] gathered src feats
    const float*  __restrict__ r,       // [E, KEDGE] relu(edge MLP1)
    const int*    __restrict__ tgt,     // [E]
    float*        __restrict__ aggr)    // [N_NODES, M_OUT], atomic accum
{
  constexpr int NTILES = M_OUT / 16;
  constexpr int TPW    = NTILES / 4;                    // tiles per wave
  constexpr int NCHUNK = (M_IN_PAD * KEDGE) / 32;
  constexpr int CPS    = NCHUNK / KSPLIT;

  __shared__ float s_x[16][M_IN_PAD + 4];
  __shared__ float s_r[16][KEDGE + 4];
  __shared__ int   s_tgt[16];

  const int e0   = blockIdx.x * 16;
  const int tid  = threadIdx.x;
  const int wave = tid >> 5;
  const int lane = tid & 31;

  for (int idx = tid; idx < 16 * M_IN_PAD; idx += 128) {
    const int rr = idx / M_IN_PAD, cc = idx % M_IN_PAD;
    s_x[rr][cc] = xs[(size_t)(e0 + rr) * M_IN_PAD + cc];
  }
  for (int idx = tid; idx < 16 * KEDGE; idx += 128) {
    const int rr = idx / KEDGE, cc = idx % KEDGE;
    s_r[rr][cc] = r[(size_t)(e0 + rr) * KEDGE + cc];
  }
  if (tid < 16) s_tgt[tid] = tgt[e0 + tid];
  __syncthreads();

  const int row = lane & 15;
  const int sel = lane >> 4;

  v8f acc[TPW];
#pragma unroll
  for (int t = 0; t < TPW; ++t) acc[t] = v8f{0.f, 0.f, 0.f, 0.f, 0.f, 0.f, 0.f, 0.f};

  const int c0 = blockIdx.y * CPS;
#pragma unroll 2
  for (int c = c0; c < c0 + CPS; ++c) {
    const int p0 = c * 32;
    const int k  = p0 >> LG_MIN;           // constant over the 32-chunk
    const int i0 = p0 & (M_IN_PAD - 1);
    const float rv = s_r[row][k];

    // Unguarded speculative prefetch of the next chunk's B tile (ISA: spec.
    // prefetch translation failures are silently dropped; one-chunk over-read
    // stays inside the workspace). Keeps the hot loop branch-free.
    __builtin_prefetch(
        Bfrag + ((size_t)((c + 1) * NTILES + wave * TPW) * 32 + lane) * 16, 0, 1);

    v16bf a;
#pragma unroll
    for (int h = 0; h < 16; ++h) {
      const int kk = (h & 7) + (sel << 3) + ((h >> 3) << 4);
      a[h] = (bf16_t)(s_x[row][i0 + kk] * rv);
    }

#pragma unroll
    for (int t = 0; t < TPW; ++t) {
      const int tile = wave * TPW + t;
      const v16bf b =
          *(const v16bf*)(Bfrag + ((size_t)(c * NTILES + tile) * 32 + lane) * 16);
      acc[t] = __builtin_amdgcn_wmma_f32_16x16x32_bf16(
          false, a, false, b, (short)0, acc[t], false, false);
    }
  }

  if (blockIdx.y == 0) {
    // fold in Σ_i xs[e,i]·b2[i,o] using the same WMMA machinery
    constexpr int CB = M_IN_PAD / 32;
    for (int c = 0; c < CB; ++c) {
      const int i0 = c * 32;
      v16bf a;
#pragma unroll
      for (int h = 0; h < 16; ++h) {
        const int kk = (h & 7) + (sel << 3) + ((h >> 3) << 4);
        a[h] = (bf16_t)(s_x[row][i0 + kk]);
      }
#pragma unroll
      for (int t = 0; t < TPW; ++t) {
        const int tile = wave * TPW + t;
        const v16bf b =
            *(const v16bf*)(b2frag + ((size_t)(c * NTILES + tile) * 32 + lane) * 16);
        acc[t] = __builtin_amdgcn_wmma_f32_16x16x32_bf16(
            false, a, false, b, (short)0, acc[t], false, false);
      }
    }
  }

  // Epilogue: C/D layout -> edge m = j + 8*sel, col o = tile*16 + (lane&15).
#pragma unroll
  for (int t = 0; t < TPW; ++t) {
    const int o = (wave * TPW + t) * 16 + row;
#pragma unroll
    for (int j = 0; j < 8; ++j) {
      const int m = j + (sel << 3);
      float* dst = &aggr[(size_t)s_tgt[m] * M_OUT + o];
      (void)__hip_atomic_fetch_add(dst, acc[t][j], __ATOMIC_RELAXED,
                                   __HIP_MEMORY_SCOPE_AGENT);
    }
  }
}

// ---------------------------------------------------------------------------
// h_out = elu(aggr + h_in @ root + bias)    (tiny: <=134 MFLOP)
// ---------------------------------------------------------------------------
__global__ __launch_bounds__(256) void node_update_kernel(
    const float* __restrict__ aggr, const float* __restrict__ hin,
    const float* __restrict__ root, const float* __restrict__ bias,
    float* __restrict__ hout, int N, int Din, int Dout)
{
  const int t = blockIdx.x * blockDim.x + threadIdx.x;
  if (t >= N * Dout) return;
  const int n = t / Dout, o = t % Dout;
  float acc = aggr[t] + bias[o];
  for (int i = 0; i < Din; ++i)
    acc = fmaf(hin[(size_t)n * Din + i], root[(size_t)i * Dout + o], acc);
  hout[t] = acc > 0.f ? acc : (expf(acc) - 1.f);
}

__global__ __launch_bounds__(256) void pool_sum_kernel(
    const float* __restrict__ h, const int* __restrict__ batch,
    float* __restrict__ gsum, float* __restrict__ cnt, int N, int D)
{
  const int t = blockIdx.x * blockDim.x + threadIdx.x;
  if (t >= N * D) return;
  const int n = t / D, o = t % D;
  const int g = batch[n];
  (void)__hip_atomic_fetch_add(&gsum[(size_t)g * D + o], h[t], __ATOMIC_RELAXED,
                               __HIP_MEMORY_SCOPE_AGENT);
  if (o == 0)
    (void)__hip_atomic_fetch_add(&cnt[g], 1.f, __ATOMIC_RELAXED,
                                 __HIP_MEMORY_SCOPE_AGENT);
}

// FC head: one block per graph (2.6 MFLOP total — negligible).
__global__ __launch_bounds__(128) void head_kernel(
    const float* __restrict__ gsum, const float* __restrict__ cnt,
    const float* __restrict__ w1, const float* __restrict__ b1,
    const float* __restrict__ w2, const float* __restrict__ b2,
    const float* __restrict__ w3, const float* __restrict__ b3,
    float* __restrict__ out)
{
  __shared__ float sg[256], sh1[128], sh2[64];
  const int g = blockIdx.x, t = threadIdx.x;
  const float inv = 1.f / fmaxf(cnt[g], 1.f);
  for (int c = t; c < 256; c += 128) sg[c] = gsum[(size_t)g * 256 + c] * inv;
  __syncthreads();
  if (t < 128) {
    float a = b1[t];
    for (int i = 0; i < 256; ++i) a = fmaf(sg[i], w1[i * 128 + t], a);
    sh1[t] = a > 0.f ? a : (expf(a) - 1.f);
  }
  __syncthreads();
  if (t < 64) {
    float a = b2[t];
    for (int i = 0; i < 128; ++i) a = fmaf(sh1[i], w2[i * 64 + t], a);
    sh2[t] = a > 0.f ? a : (expf(a) - 1.f);
  }
  __syncthreads();
  if (t == 0) {
    float a = b3[0];
    for (int i = 0; i < 64; ++i) a = fmaf(sh2[i], w3[i], a);
    out[g] = a;
  }
}

// ---------------------------------------------------------------------------
extern "C" void kernel_launch(void* const* d_in, const int* in_sizes, int n_in,
                              void* d_out, int out_size, void* d_ws, size_t ws_size,
                              hipStream_t stream)
{
  (void)in_sizes; (void)n_in; (void)out_size; (void)ws_size;
  const int E = N_EDGES, NN = N_NODES, NG = N_GRAPHS;

  const float* x     = (const float*)d_in[0];
  const int*   eidx  = (const int*)d_in[1];
  const float* ea    = (const float*)d_in[2];
  const int*   batch = (const int*)d_in[3];
  const float* c1_w1 = (const float*)d_in[4];  const float* c1_b1 = (const float*)d_in[5];
  const float* c1_w2 = (const float*)d_in[6];  const float* c1_b2 = (const float*)d_in[7];
  const float* c1_rt = (const float*)d_in[8];  const float* c1_bs = (const float*)d_in[9];
  const float* c2_w1 = (const float*)d_in[10]; const float* c2_b1 = (const float*)d_in[11];
  const float* c2_w2 = (const float*)d_in[12]; const float* c2_b2 = (const float*)d_in[13];
  const float* c2_rt = (const float*)d_in[14]; const float* c2_bs = (const float*)d_in[15];
  const float* c3_w1 = (const float*)d_in[16]; const float* c3_b1 = (const float*)d_in[17];
  const float* c3_w2 = (const float*)d_in[18]; const float* c3_b2 = (const float*)d_in[19];
  const float* c3_rt = (const float*)d_in[20]; const float* c3_bs = (const float*)d_in[21];
  const float* fc1_w = (const float*)d_in[22]; const float* fc1_b = (const float*)d_in[23];
  const float* fc2_w = (const float*)d_in[24]; const float* fc2_b = (const float*)d_in[25];
  const float* fc3_w = (const float*)d_in[26]; const float* fc3_b = (const float*)d_in[27];

  const int* src = eidx;
  const int* tgt = eidx + E;

  char* wsp = (char*)d_ws;
  size_t off = 0;
  auto alloc = [&](size_t bytes) -> void* {
    off = (off + 255) & ~(size_t)255;
    void* p = wsp + off;
    off += bytes;
    return p;
  };

  float* r1  = (float*)alloc((size_t)E * 512 * 4);
  float* r2  = (float*)alloc((size_t)E * 128 * 4);
  float* r3  = (float*)alloc((size_t)E * 128 * 4);
  float* xs1 = (float*)alloc((size_t)E * 64 * 4);
  float* xs2 = (float*)alloc((size_t)E * 128 * 4);
  float* xs3 = (float*)alloc((size_t)E * 256 * 4);
  float* h1  = (float*)alloc((size_t)NN * 128 * 4);
  float* h2  = (float*)alloc((size_t)NN * 256 * 4);
  float* h3  = (float*)alloc((size_t)NN * 256 * 4);
  float* ag1 = (float*)alloc((size_t)NN * 128 * 4);
  float* ag2 = (float*)alloc((size_t)NN * 256 * 4);
  float* ag3 = (float*)alloc((size_t)NN * 256 * 4);
  bf16_t* Bf1 = (bf16_t*)alloc((size_t)64 * 512 * 128 * 2);
  bf16_t* Bf2 = (bf16_t*)alloc((size_t)128 * 128 * 256 * 2);
  bf16_t* Bf3 = (bf16_t*)alloc((size_t)256 * 128 * 256 * 2);
  bf16_t* bb1 = (bf16_t*)alloc((size_t)64 * 128 * 2);
  bf16_t* bb2 = (bf16_t*)alloc((size_t)128 * 256 * 2);
  bf16_t* bb3 = (bf16_t*)alloc((size_t)256 * 256 * 2);
  float* gsum = (float*)alloc((size_t)NG * 256 * 4);
  float* cnt  = (float*)alloc((size_t)NG * 4);

  // -- weight pre-swizzle into WMMA fragment order (runs every call) --
  swizzle_w2_kernel<6, 64, 37, 512, 128><<<2048, 256, 0, stream>>>(c1_w2, Bf1);
  swizzle_w2_kernel<7, 128, 128, 128, 256><<<2048, 256, 0, stream>>>(c2_w2, Bf2);
  swizzle_w2_kernel<8, 256, 256, 128, 256><<<4096, 256, 0, stream>>>(c3_w2, Bf3);
  swizzle_b2_kernel<64, 37, 128><<<16, 256, 0, stream>>>(c1_b2, bb1);
  swizzle_b2_kernel<128, 128, 256><<<64, 256, 0, stream>>>(c2_b2, bb2);
  swizzle_b2_kernel<256, 256, 256><<<128, 256, 0, stream>>>(c3_b2, bb3);

  // -- edge MLP first layers (shared input edge_attr) --
  edge_mlp1_kernel<<<(E * 512 + 255) / 256, 256, 0, stream>>>(ea, c1_w1, c1_b1, r1, E, 512);
  edge_mlp1_kernel<<<(E * 128 + 255) / 256, 256, 0, stream>>>(ea, c2_w1, c2_b1, r2, E, 128);
  edge_mlp1_kernel<<<(E * 128 + 255) / 256, 256, 0, stream>>>(ea, c3_w1, c3_b1, r3, E, 128);

  // -------- conv1: 37(->64) -> 128 --------
  gather_pad_kernel<<<(E * 64 + 255) / 256, 256, 0, stream>>>(x, src, xs1, E, 37, 64);
  zero_kernel<<<(NN * 128 + 255) / 256, 256, 0, stream>>>(ag1, (long)NN * 128);
  zgemm_scatter_kernel<6, 64, 512, 128, 8>
      <<<dim3(E / 16, 8), 128, 0, stream>>>(Bf1, bb1, xs1, r1, tgt, ag1);
  node_update_kernel<<<(NN * 128 + 255) / 256, 256, 0, stream>>>(
      ag1, x, c1_rt, c1_bs, h1, NN, 37, 128);

  // -------- conv2: 128 -> 256 --------
  gather_pad_kernel<<<(E * 128 + 255) / 256, 256, 0, stream>>>(h1, src, xs2, E, 128, 128);
  zero_kernel<<<(NN * 256 + 255) / 256, 256, 0, stream>>>(ag2, (long)NN * 256);
  zgemm_scatter_kernel<7, 128, 128, 256, 8>
      <<<dim3(E / 16, 8), 128, 0, stream>>>(Bf2, bb2, xs2, r2, tgt, ag2);
  node_update_kernel<<<(NN * 256 + 255) / 256, 256, 0, stream>>>(
      ag2, h1, c2_rt, c2_bs, h2, NN, 128, 256);

  // -------- conv3: 256 -> 256 --------
  gather_pad_kernel<<<(E * 256 + 255) / 256, 256, 0, stream>>>(h2, src, xs3, E, 256, 256);
  zero_kernel<<<(NN * 256 + 255) / 256, 256, 0, stream>>>(ag3, (long)NN * 256);
  zgemm_scatter_kernel<8, 256, 128, 256, 8>
      <<<dim3(E / 16, 8), 128, 0, stream>>>(Bf3, bb3, xs3, r3, tgt, ag3);
  node_update_kernel<<<(NN * 256 + 255) / 256, 256, 0, stream>>>(
      ag3, h2, c3_rt, c3_bs, h3, NN, 256, 256);

  // -------- global mean pool + FC head --------
  zero_kernel<<<(NG * 256 + 255) / 256, 256, 0, stream>>>(gsum, (long)NG * 256);
  zero_kernel<<<1, 64, 0, stream>>>(cnt, (long)NG);
  pool_sum_kernel<<<(NN * 256 + 255) / 256, 256, 0, stream>>>(h3, batch, gsum, cnt, NN, 256);
  head_kernel<<<NG, 128, 0, stream>>>(gsum, cnt, fc1_w, fc1_b, fc2_w, fc2_b,
                                      fc3_w, fc3_b, (float*)d_out);
}
